// _NonLocalBlockND_1116691497558
// MI455X (gfx1250) — matrix operand order repaired and checked by reading, not compiled, not run on hardware
//
#include <hip/hip_runtime.h>

typedef __attribute__((ext_vector_type(16))) __bf16 v16bf;
typedef __attribute__((ext_vector_type(8)))  float  v8f;

union FragU { uint4 q[2]; v16bf v; };

__device__ __forceinline__ unsigned short f2us(float f) {
  union { float f; unsigned u; } v; v.f = f;
  return (unsigned short)(v.u >> 16);              // bf16 truncate
}
__device__ __forceinline__ float us2f(unsigned short s) {
  union { unsigned u; float f; } v; v.u = ((unsigned)s) << 16;
  return v.f;
}
// A fragment (16-bit A 16x32 layout): 8 elems at p, 8 elems at p+16
__device__ __forceinline__ v16bf ldfragA(const unsigned short* p) {
  FragU f;
  f.q[0] = *(const uint4*)(p);
  f.q[1] = *(const uint4*)(p + 16);
  return f.v;
}
// B fragment (16-bit B 32x16 layout): 8 elems at p, 8 elems at p+16
__device__ __forceinline__ v16bf ldfragB(const unsigned short* p) {
  FragU f;
  f.q[0] = *(const uint4*)(p);
  f.q[1] = *(const uint4*)(p + 16);
  return f.v;
}
// pack 8 f32 accumulators -> 8 consecutive bf16 (one b128 store)
__device__ __forceinline__ uint4 pack8(const v8f& a) {
  uint4 u;
  u.x = (unsigned)f2us(a[0]) | ((unsigned)f2us(a[1]) << 16);
  u.y = (unsigned)f2us(a[2]) | ((unsigned)f2us(a[3]) << 16);
  u.z = (unsigned)f2us(a[4]) | ((unsigned)f2us(a[5]) << 16);
  u.w = (unsigned)f2us(a[6]) | ((unsigned)f2us(a[7]) << 16);
  return u;
}

// ---- async global->LDS staging (CDNA5 GLOBAL_LOAD_ASYNC_TO_LDS_B128) ------
#if defined(__has_builtin)
# if __has_builtin(__builtin_amdgcn_global_load_async_to_lds_b128) && \
     __has_builtin(__builtin_amdgcn_s_wait_asynccnt)
#  define USE_ASYNC_LDS 1
# endif
#endif

#ifdef USE_ASYNC_LDS
typedef __attribute__((vector_size(16))) int v4i_t;
typedef __attribute__((address_space(1))) v4i_t gas_v4i;   // global int4
typedef __attribute__((address_space(3))) v4i_t las_v4i;   // LDS int4
__device__ __forceinline__ void stage16(void* lds, const void* g) {
  __builtin_amdgcn_global_load_async_to_lds_b128(
      (gas_v4i*)(v4i_t*)g, (las_v4i*)(v4i_t*)lds, 0, 0);
}
__device__ __forceinline__ void stage_wait() {
  __builtin_amdgcn_s_wait_asynccnt(0);
}
#else
__device__ __forceinline__ void stage16(void* lds, const void* g) {
  *(uint4*)lds = *(const uint4*)g;
}
__device__ __forceinline__ void stage_wait() {}
#endif

// ---------------------------------------------------------------------------
// Kernel 0: convert all 1x1-conv weights fp32 -> bf16 once (256KB total)
// ---------------------------------------------------------------------------
__global__ __launch_bounds__(256) void wconvert_kernel(
    const float* __restrict__ gw, const float* __restrict__ tw,
    const float* __restrict__ pw, const float* __restrict__ ww,
    unsigned short* __restrict__ o)
{
  int i = blockIdx.x * 256 + threadIdx.x;          // 131072 total
  const float* srcs[4] = {gw, tw, pw, ww};
  int s = i >> 15, j = i & 32767;
  o[i] = f2us(srcs[s][j]);
}

// ---------------------------------------------------------------------------
// Kernel 1: fused 1x1-conv projections (g, theta, phi) + patchify
// ---------------------------------------------------------------------------
__global__ __launch_bounds__(256) void proj_kernel(
    const float* __restrict__ x,
    const unsigned short* __restrict__ wbf,        // bf16 weights, 3x(128x256)
    const float* __restrict__ gb, const float* __restrict__ tb,
    const float* __restrict__ pb,
    unsigned short* __restrict__ outgT,            // [B][8192 d][256 m]
    unsigned short* __restrict__ outt,             // [B][256 n][8192 d]
    unsigned short* __restrict__ outp)
{
  extern __shared__ unsigned short xs[];           // [128 pix][256 ch] = 64KB
  const int tid  = threadIdx.x;
  const int p0   = blockIdx.x * 128;               // one image row
  const int b    = p0 >> 14;
  const int rem0 = p0 & 16383;

  for (int idx = tid; idx < 8192; idx += 256) {    // float4-coalesced stage
    int c = idx >> 5, pq = idx & 31;
    float4 v = *(const float4*)(x + ((size_t)(b * 256 + c) << 14) + rem0 + pq * 4);
    int base = (pq * 4) * 256 + c;
    xs[base]       = f2us(v.x);
    xs[base + 256] = f2us(v.y);
    xs[base + 512] = f2us(v.z);
    xs[base + 768] = f2us(v.w);
  }
  __syncthreads();

  const int wid = tid >> 5, lane = tid & 31;
  const int row = lane & 15, hi = lane >> 4;
  const int kbA = hi * 8, kbB = hi * 8;
  const unsigned short* arow = xs + (wid * 16 + row) * 256;

  // store geometry: lane's 8 D values = 8 consecutive pixels (one patch row)
  const int m0   = wid * 16 + hi * 8;
  const int rem  = rem0 + m0;
  const int h    = rem >> 7, w0 = rem & 127;
  const int n    = ((h >> 3) << 4) + (w0 >> 3);
  const int doff = ((h & 7) << 3);                 // + och*64 later

  const float* bsets[3] = {gb, tb, pb};
  unsigned short* osets[3] = {outgT, outt, outp};

  for (int s = 0; s < 3; ++s) {
    const float* bv = bsets[s];
    unsigned short* om = osets[s];
    const unsigned short* wmat = wbf + s * 32768;
    for (int nt = 0; nt < 8; ++nt) {
      const int och = nt * 16 + row;
      const unsigned short* brow = wmat + och * 256;
      const float bias = bv[och];
      v8f acc;
      #pragma unroll
      for (int r = 0; r < 8; ++r) acc[r] = bias;
      #pragma unroll
      for (int kt = 0; kt < 8; ++kt) {             // K = 256
        const int k0 = kt * 32;
        v16bf a  = ldfragA(arow + k0 + kbA);
        v16bf bm = ldfragB(brow + k0 + kbB);
        acc = __builtin_amdgcn_wmma_f32_16x16x32_bf16(
                  false, a, false, bm, (short)0, acc, false, false);
      }
      if (s == 0) {                                // gT[b][d][m]: m contiguous
        #pragma unroll
        for (int r = 0; r < 8; ++r) {
          int d = och * 64 + doff + r;
          om[((size_t)b << 21) + ((size_t)d << 8) + n] = f2us(acc[r]);
        }
      } else {                                     // tokens: d contiguous
        int d0 = och * 64 + doff;
        *(uint4*)(om + ((size_t)b << 21) + ((size_t)n << 13) + d0) = pack8(acc);
      }
    }
  }
}

// ---------------------------------------------------------------------------
// Kernel 2: attention.  One block per (batch, 64-row group), 16 waves.
// ---------------------------------------------------------------------------
__global__ __launch_bounds__(512) void attn_kernel(
    const unsigned short* __restrict__ theta,
    const unsigned short* __restrict__ phi,
    const unsigned short* __restrict__ gT,         // [B][8192 d][256 m]
    unsigned short* __restrict__ y2)               // [B][256 n][64 ij][128 ci]
{
  extern __shared__ unsigned char smem[];
  float* f            = (float*)smem;                       // 64x256 f32 (A)
  unsigned short* gS  = (unsigned short*)smem;              // 64x256 bf16 (A, ph3)
  unsigned short* thS = (unsigned short*)(smem + 65536);    // 64x64 bf16 (B)
  unsigned short* phS = thS + 64 * 64;                      // 256x64 bf16 (B)
  unsigned short* fbf = (unsigned short*)(smem + 65536);    // 64x256 bf16 (B, ph3)

  const int tid = threadIdx.x;
  const int b   = blockIdx.x >> 2;
  const int n0  = (blockIdx.x & 3) * 64;
  const size_t bbase = (size_t)b << 21;

  const int wid = tid >> 5, lane = tid & 31;
  const int row = lane & 15, hi = lane >> 4;
  const int kbA = hi * 8, kbB = hi * 8;

  // ---- phase 1: f = theta . phi^T, K = 8192 in 64-wide LDS chunks ----
  v8f acc[4];
  #pragma unroll
  for (int i = 0; i < 4; ++i) {
    #pragma unroll
    for (int r = 0; r < 8; ++r) acc[i][r] = 0.f;
  }

  for (int kc = 0; kc < 8192; kc += 64) {
    {                                              // thS: 512 x16B, 1/thread
      int r = tid >> 3, kq = tid & 7;
      stage16(thS + r * 64 + kq * 8,
              theta + bbase + ((size_t)(n0 + r) << 13) + kc + kq * 8);
    }
    for (int idx = tid; idx < 2048; idx += 512) {  // phS: 2048 x16B
      int r = idx >> 3, kq = idx & 7;
      stage16(phS + r * 64 + kq * 8,
              phi + bbase + ((size_t)r << 13) + kc + kq * 8);
    }
    if (kc + 64 < 8192)                            // global_prefetch_b8
      __builtin_prefetch(phi + bbase + kc + 64, 0, 0);
    stage_wait();
    __syncthreads();
    #pragma unroll
    for (int i = 0; i < 4; ++i) {                  // 64 tiles / 16 waves
      int T = wid + (i << 4);
      int nt = T & 3, mt = T >> 2;
      const unsigned short* ap = thS + (nt * 16 + row) * 64;
      const unsigned short* bp = phS + (mt * 16 + row) * 64;
      #pragma unroll
      for (int kk = 0; kk < 64; kk += 32) {
        v16bf a  = ldfragA(ap + kk + kbA);
        v16bf bm = ldfragB(bp + kk + kbB);
        acc[i] = __builtin_amdgcn_wmma_f32_16x16x32_bf16(
                     false, a, false, bm, (short)0, acc[i], false, false);
      }
    }
    __syncthreads();
  }

  #pragma unroll
  for (int i = 0; i < 4; ++i) {
    int T = wid + (i << 4);
    int nt = T & 3, mt = T >> 2;
    #pragma unroll
    for (int r = 0; r < 8; ++r)
      f[((nt * 16 + r + hi * 8) << 8) + mt * 16 + row] = acc[i][r];
  }
  __syncthreads();

  // ---- phase 2: row softmax ----
  if (tid < 64) {
    float* fr = f + (tid << 8);
    float mx = fr[0];
    for (int m = 1; m < 256; ++m) mx = fmaxf(mx, fr[m]);
    float sum = 0.f;
    for (int m = 0; m < 256; ++m) { float e = __expf(fr[m] - mx); fr[m] = e; sum += e; }
    float inv = 1.0f / sum;
    for (int m = 0; m < 256; ++m) fr[m] *= inv;
  }
  __syncthreads();
  for (int idx = tid; idx < 64 * 256; idx += 512)  // bf16 copy for A frags
    fbf[idx] = f2us(f[idx]);
  __syncthreads();

  // ---- phase 3: y = softmax(f) . g, 64-column d-chunks ----
  const int dt = wid & 3, ns = wid >> 2;
  const unsigned short* ap = fbf + (ns * 16 + row) * 256;
  for (int dc = 0; dc < 8192; dc += 64) {
    for (int idx = tid; idx < 2048; idx += 512) {  // gS: 64x256 = 2048 x16B
      int d = idx >> 5, q = idx & 31;
      stage16(gS + d * 256 + q * 8,
              gT + bbase + ((size_t)(dc + d) << 8) + q * 8);
    }
    if (dc + 64 < 8192)
      __builtin_prefetch(gT + bbase + ((size_t)(dc + 64) << 8), 0, 0);
    stage_wait();
    __syncthreads();

    const unsigned short* bp = gS + (dt * 16 + row) * 256;
    v8f yacc;
    #pragma unroll
    for (int r = 0; r < 8; ++r) yacc[r] = 0.f;
    #pragma unroll
    for (int kk = 0; kk < 256; kk += 32) {
      v16bf a  = ldfragA(ap + kk + kbA);
      v16bf bm = ldfragB(bp + kk + kbB);
      yacc = __builtin_amdgcn_wmma_f32_16x16x32_bf16(
                 false, a, false, bm, (short)0, yacc, false, false);
    }
    int d  = dc + dt * 16 + row;
    int ci = d >> 6, ij = d & 63;
    #pragma unroll
    for (int r = 0; r < 8; ++r) {
      int nr = n0 + ns * 16 + r + hi * 8;
      y2[bbase + ((size_t)nr << 13) + (ij << 7) + ci] = f2us(yacc[r]);
    }
    __syncthreads();                               // before gS is restaged
  }
}

// ---------------------------------------------------------------------------
// Kernel 3: output 1x1 conv (K=128) + unpatchify + BN-stat accumulation
// ---------------------------------------------------------------------------
__global__ __launch_bounds__(256) void outconv_kernel(
    const unsigned short* __restrict__ y2,         // [B][n][ij][ci]
    const unsigned short* __restrict__ wwbf,       // bf16 [256][128]
    const float* __restrict__ wb,
    unsigned short* __restrict__ wy,
    float* __restrict__ gsum, float* __restrict__ gsq)
{
  __shared__ unsigned short ys[128 * 128];         // [pix][ci], 32KB
  __shared__ float bsum[256], bsq[256];
  const int tid  = threadIdx.x;
  const int p0   = blockIdx.x * 128;
  const int b    = p0 >> 14;
  const int rem0 = p0 & 16383;
  const int h    = rem0 >> 7;                      // strip == one image row
  const size_t bbase = (size_t)b << 21;

  bsum[tid] = 0.f; bsq[tid] = 0.f;
  for (int idx = tid; idx < 2048; idx += 256) {    // 2048 x16B
    int pix = idx >> 4, cq = idx & 15;
    int n  = ((h >> 3) << 4) + (pix >> 3);
    int ij = ((h & 7) << 3) + (pix & 7);
    stage16(ys + pix * 128 + cq * 8,
            y2 + bbase + ((size_t)n << 13) + (ij << 7) + cq * 8);
  }
  stage_wait();
  __syncthreads();

  const int wid = tid >> 5, lane = tid & 31;
  const int row = lane & 15, hi = lane >> 4;
  const int kbA = hi * 8, kbB = hi * 8;
  const unsigned short* arow = ys + (wid * 16 + row) * 128;
  const int m0 = wid * 16 + hi * 8;                // 8 consecutive pixels

  for (int nt = 0; nt < 16; ++nt) {
    const int och = nt * 16 + row;
    const unsigned short* brow = wwbf + och * 128;
    const float bias = wb[och];
    v8f acc;
    #pragma unroll
    for (int r = 0; r < 8; ++r) acc[r] = bias;
    #pragma unroll
    for (int kt = 0; kt < 4; ++kt) {               // K = 128
      const int k0 = kt * 32;
      v16bf a  = ldfragA(arow + k0 + kbA);
      v16bf bm = ldfragB(brow + k0 + kbB);
      acc = __builtin_amdgcn_wmma_f32_16x16x32_bf16(
                false, a, false, bm, (short)0, acc, false, false);
    }
    *(uint4*)(wy + ((size_t)(b * 256 + och) << 14) + rem0 + m0) = pack8(acc);
    float ls = 0.f, lq = 0.f;
    #pragma unroll
    for (int r = 0; r < 8; ++r) { float v = acc[r]; ls += v; lq += v * v; }
    atomicAdd(&bsum[och], ls);
    atomicAdd(&bsq[och], lq);
  }
  __syncthreads();
  atomicAdd(&gsum[tid], bsum[tid]);
  atomicAdd(&gsq[tid], bsq[tid]);
}

// ---------------------------------------------------------------------------
// Small kernels: stat init, BN finalize, elementwise apply + residual
// ---------------------------------------------------------------------------
__global__ void zero_stats(float* gsum, float* gsq) {
  int t = threadIdx.x;
  gsum[t] = 0.f; gsq[t] = 0.f;
}

__global__ void bn_stats(const float* __restrict__ gsum, const float* __restrict__ gsq,
                         const float* __restrict__ gamma, const float* __restrict__ beta,
                         float* __restrict__ scale, float* __restrict__ shift) {
  int o = threadIdx.x;
  const float inv_cnt = 1.0f / 131072.0f;          // B*H*W
  float mean = gsum[o] * inv_cnt;
  float var  = gsq[o] * inv_cnt - mean * mean;
  float sc   = gamma[o] * rsqrtf(var + 1e-5f);
  scale[o] = sc;
  shift[o] = beta[o] - mean * sc;
}

__global__ __launch_bounds__(256) void apply_kernel(
    const unsigned short* __restrict__ wy,
    const float* __restrict__ scale, const float* __restrict__ shift,
    const float* __restrict__ x, float* __restrict__ out)
{
  size_t i = ((size_t)blockIdx.x * 256 + threadIdx.x) * 4;
  int c = (int)((i >> 14) & 255);
  float sc = scale[c], sh = shift[c];
  uint2 w2  = *(const uint2*)(wy + i);             // 4 bf16
  float4 xv = *(const float4*)(x + i);
  float4 o;
  o.x = us2f((unsigned short)(w2.x & 0xffffu)) * sc + sh + xv.x;
  o.y = us2f((unsigned short)(w2.x >> 16))     * sc + sh + xv.y;
  o.z = us2f((unsigned short)(w2.y & 0xffffu)) * sc + sh + xv.z;
  o.w = us2f((unsigned short)(w2.y >> 16))     * sc + sh + xv.w;
  *(float4*)(out + i) = o;
}

// ---------------------------------------------------------------------------
extern "C" void kernel_launch(void* const* d_in, const int* in_sizes, int n_in,
                              void* d_out, int out_size, void* d_ws, size_t ws_size,
                              hipStream_t stream) {
  const float* x     = (const float*)d_in[0];
  const float* g_w   = (const float*)d_in[1];
  const float* g_b   = (const float*)d_in[2];
  const float* th_w  = (const float*)d_in[3];
  const float* th_b  = (const float*)d_in[4];
  const float* ph_w  = (const float*)d_in[5];
  const float* ph_b  = (const float*)d_in[6];
  const float* w_w   = (const float*)d_in[7];
  const float* w_b   = (const float*)d_in[8];
  const float* gamma = (const float*)d_in[9];
  const float* beta  = (const float*)d_in[10];
  float* out = (float*)d_out;

  const size_t PE = 16777216;                      // 8*256*8192 bf16 elems
  unsigned short* wsu = (unsigned short*)d_ws;
  unsigned short* pgT = wsu;                       // g transposed [b][d][m]
  unsigned short* pt  = wsu + PE;                  // theta tokens
  unsigned short* pp  = wsu + 2 * PE;              // phi tokens
  unsigned short* py2 = wsu + 3 * PE;              // y2 [b][n][ij][ci]
  unsigned short* pwy = wsu + 4 * PE;              // w_y [B,C,H,W] bf16
  unsigned short* pwbf = wsu + 4 * PE + 33554432;  // bf16 weights (131072)
  float* stats = (float*)(pwbf + 131072);
  float* gsum  = stats;
  float* gsq   = stats + 256;
  float* scale = stats + 512;
  float* shift = stats + 768;

  wconvert_kernel<<<512, 256, 0, stream>>>(g_w, th_w, ph_w, w_w, pwbf);

  proj_kernel<<<1024, 256, 128 * 256 * 2, stream>>>(
      x, pwbf, g_b, th_b, ph_b, pgT, pt, pp);

  zero_stats<<<1, 256, 0, stream>>>(gsum, gsq);

  // LDS: region A 64KB (f | gS) + region B 40KB (thS+phS | fbf)
  const size_t attn_lds = 65536 + 64 * 64 * 2 + 256 * 64 * 2; // 106,496 B
  attn_kernel<<<32, 512, attn_lds, stream>>>(pt, pp, pgT, py2);

  outconv_kernel<<<1024, 256, 0, stream>>>(py2, pwbf + 98304, w_b, pwy, gsum, gsq);

  bn_stats<<<1, 256, 0, stream>>>(gsum, gsq, gamma, beta, scale, shift);

  apply_kernel<<<32768, 256, 0, stream>>>(pwy, scale, shift, x, out);
}
